// Mul_Window_attention_67345087201623
// MI455X (gfx1250) — matrix operand order repaired and checked by reading, not compiled, not run on hardware
//
#include <hip/hip_runtime.h>

typedef __bf16 v16bf __attribute__((ext_vector_type(16)));
typedef float  v8f   __attribute__((ext_vector_type(8)));
typedef float  v4f   __attribute__((ext_vector_type(4)));
typedef unsigned int v4u __attribute__((ext_vector_type(4)));
typedef int v4i __attribute__((ext_vector_type(4)));
typedef int v8i __attribute__((ext_vector_type(8)));
typedef unsigned short u16;

#define DIMC  256
#define HWSZ  16384
#define BATCH 8
#define WIDTH 128
#define HD    64
#define QSCALE 0.125f   // 64^-0.5

__device__ __forceinline__ u16 f2bf(float f) {
  union { float f; unsigned u; } a; a.f = f;
  unsigned r = a.u + 0x7fffu + ((a.u >> 16) & 1u);
  return (u16)(r >> 16);
}
__device__ __forceinline__ float bf2f(u16 h) {
  union { unsigned u; float f; } a; a.u = ((unsigned)h) << 16;
  return a.f;
}

union FragU { v4u q[2]; u16 u[16]; v16bf v; };

// A-matrix 16x32 bf16: lane holds row M=lane&15; half h=lane>>4 holds
// K chunks [h*8, h*8+8) and [16+h*8, 16+h*8+8)   (ISA 7.12.2)
__device__ __forceinline__ v16bf load_frag_a(const u16* row, int h) {
  FragU f;
  f.q[0] = *(const v4u*)(row + h * 8);
  f.q[1] = *(const v4u*)(row + 16 + h * 8);
  return f.v;
}
// B-matrix 32x16 bf16: lane holds col N=lane&15; half h holds K chunks
// [h*8, h*8+8) and [16+h*8, 16+h*8+8)
__device__ __forceinline__ v16bf load_frag_b(const u16* row, int h) {
  FragU f;
  f.q[0] = *(const v4u*)(row + h * 8);
  f.q[1] = *(const v4u*)(row + 16 + h * 8);
  return f.v;
}
__device__ __forceinline__ v8f wmma_bf16(v16bf a, v16bf b, v8f c) {
  return __builtin_amdgcn_wmma_f32_16x16x32_bf16(false, a, false, b, (short)0, c,
                                                 false, false);
}

// ---------------- Tensor Data Mover: 3-D V tile -> LDS ------------------------
// Tile: x = WS contiguous u16, y = WS rows (stride 128), z = 64 d-planes
// (stride 16384).  Result in LDS: [d][token] contiguous == vt layout.
// PAD: after every 8 DWORDs stored (16 tokens, one d-row for WS=4), skip
// 8 DWORDs (16 tokens) of LDS -> NK=32 row pitch with pre-zeroed pad columns.
template <int WS, bool PAD>
__device__ __forceinline__ void tdm_load_vt(unsigned lds_off, const u16* gsrc) {
  unsigned long long ga = (unsigned long long)(const void*)gsrc;
  v4i g0;
  g0[0] = 1;                                         // count=1, user descriptor
  g0[1] = (int)lds_off;                              // lds_addr (bytes)
  g0[2] = (int)(unsigned)(ga & 0xffffffffull);       // global_addr[31:0]
  g0[3] = (int)(((unsigned)(ga >> 32) & 0x01ffffffu) | (2u << 30)); // type=2
  unsigned w0 = 1u << 16;                            // data_size = 2 bytes
  if (PAD) w0 |= (1u << 20) | (2u << 22) | (7u << 25); // pad_en, int=8dw, amt=8dw
  v8i g1;
  g1[0] = (int)w0;                                   // mask=0 (not in cluster)
  g1[1] = (int)((unsigned)WS << 16);                 // tensor_dim0 lo16
  g1[2] = (int)((unsigned)WS << 16);                 // td0 hi=0 | td1 lo16
  g1[3] = (int)((unsigned)WS << 16);                 // td1 hi=0 | tile_dim0
  g1[4] = (int)((unsigned)WS | (64u << 16));         // tile_dim1 | tile_dim2
  g1[5] = 128;                                       // tensor_dim0_stride lo32
  g1[6] = (int)(16384u << 16);                       // str0 hi=0 | str1 lo16
  g1[7] = 0;                                         // str1 hi (16384>>16 = 0)
  v4i g2 = {64, 0, 0, 0};                            // tensor_dim2 = 64
  v4i g3 = {0, 0, 0, 0};
  asm volatile("tensor_load_to_lds %0, %1, %2, %3"
               :: "s"(g0), "s"(g1), "s"(g2), "s"(g3) : "memory");
}

// ---------------- x [B,C,HW] f32 -> xt [B,HW,C] bf16 (tiled transpose) --------
__global__ __launch_bounds__(256) void xpose_cvt(const float* __restrict__ x,
                                                 u16* __restrict__ xt) {
  __shared__ u16 tile[32][33];
  int b = blockIdx.z, c0 = blockIdx.y * 32, hw0 = blockIdx.x * 32;
  int tx = threadIdx.x, ty = threadIdx.y;
#pragma unroll
  for (int i = 0; i < 4; ++i) {
    int c = c0 + ty + i * 8;
    tile[ty + i * 8][tx] = f2bf(x[(b * DIMC + c) * HWSZ + hw0 + tx]);
  }
  __syncthreads();
#pragma unroll
  for (int i = 0; i < 4; ++i) {
    int hwl = ty + i * 8;
    xt[(b * HWSZ + hw0 + hwl) * DIMC + c0 + tx] = tile[tx][hwl];
  }
}

// ---------------- f32 -> bf16 elementwise (weights) ---------------------------
__global__ void cvt_bf16(const float* __restrict__ s, u16* __restrict__ d, int n) {
  int i = blockIdx.x * blockDim.x + threadIdx.x;
  if (i < n) d[i] = f2bf(s[i]);
}

// ---------------- QKV GEMM: qkv[pix, o] = sum_c xt[pix,c] * wq[o,c] -----------
__global__ __launch_bounds__(256) void qkv_gemm(const u16* __restrict__ xt,
                                                const u16* __restrict__ wq,
                                                u16* __restrict__ qkv) {
  int wave = threadIdx.x >> 5, lane = threadIdx.x & 31;
  int h = lane >> 4, ln = lane & 15;
  int o0 = blockIdx.y * 16;
  int p0 = blockIdx.x * 512 + wave * 64;
  v8f acc[4] = {};
  const u16* arow = wq + (o0 + ln) * DIMC;
  const u16* brow = xt + (p0 + ln) * DIMC;
  __builtin_prefetch(arow + 128, 0, 2);      // -> global_prefetch_b8
  __builtin_prefetch(brow + 128, 0, 2);
#pragma unroll
  for (int k0 = 0; k0 < DIMC; k0 += 32) {
    v16bf a = load_frag_a(arow + k0, h);
#pragma unroll
    for (int t = 0; t < 4; ++t) {
      v16bf b = load_frag_b(brow + (t * 16) * DIMC + k0, h);
      acc[t] = wmma_bf16(a, b, acc[t]);
    }
  }
  float scale = (o0 < DIMC) ? QSCALE : 1.0f;
#pragma unroll
  for (int t = 0; t < 4; ++t) {
    int pix = p0 + t * 16 + ln;
    union { v4u q; u16 u[8]; } st;
#pragma unroll
    for (int r = 0; r < 8; ++r) st.u[r] = f2bf(acc[t][r] * scale);
    *(v4u*)(qkv + pix * 768 + o0 + h * 8) = st.q;
  }
}

// ---------------- V -> per-head transposed global planes ----------------------
// vtg[(b*3 + head-1)*64 + d][hw] = V_head[pix, d]   (heads 1..3 use WMMA path)
__global__ __launch_bounds__(256) void vsplit(const u16* __restrict__ qkv,
                                              u16* __restrict__ vtg) {
  __shared__ u16 t[64][33];
  int hw0 = blockIdx.x * 32;
  int head = blockIdx.y + 1;
  int b = blockIdx.z;
  int voff = 2 * DIMC + head * HD;
  int tid = threadIdx.x;
#pragma unroll
  for (int it = 0; it < 8; ++it) {
    int i = it * 256 + tid;
    int d = i & 63, p = i >> 6;
    t[d][p] = qkv[(b * HWSZ + hw0 + p) * 768 + voff + d];
  }
  __syncthreads();
  int plane = (b * 3 + head - 1) * 64;
#pragma unroll
  for (int it = 0; it < 8; ++it) {
    int i = it * 256 + tid;
    int p = i & 31, d = i >> 5;
    vtg[(plane + d) * HWSZ + hw0 + p] = t[d][p];
  }
}

// ---------------- windowed attention via WMMA + TDM (ws = 4/8/16) -------------
template <int WS, int WAVES, int WPB>
__global__ __launch_bounds__(WAVES * 32) void attn_wmma(const u16* __restrict__ qkv,
                                                        const u16* __restrict__ vtg,
                                                        u16* __restrict__ y,
                                                        int head) {
  constexpr int NWIN = WS * WS;
  constexpr int NT = NWIN / 16;
  constexpr int NK = (NWIN < 32) ? 32 : NWIN;
  constexpr int WPW = WAVES / WPB;          // waves per window
  constexpr int QTPW = NT / WPW;            // q tiles per wave
  constexpr int VT_W = 64 * NK;             // u16 per window vt region
  __shared__ __align__(16) u16 smem[WPB * VT_W + WAVES * 16 * NK];

  int wave = threadIdx.x >> 5, lane = threadIdx.x & 31;
  int h = lane >> 4, ln = lane & 15;
  int winLocal = wave / WPW, wiw = wave % WPW;
  int gw = blockIdx.x * WPB + winLocal;
  int wpr = WIDTH / WS;
  int wpg = wpr * wpr;
  int b = gw / wpg, wi = gw % wpg;
  int wy = wi / wpr, wx = wi % wpr;
  int base = b * HWSZ + (wy * WS) * WIDTH + wx * WS;
  int qoff = head * HD, koff = DIMC + head * HD;

  u16* vt = smem + winLocal * VT_W;
  u16* pl = smem + WPB * VT_W + wave * 16 * NK;

  // zero padded columns (WS==4 only): vt pad stays 0 (TDM pad skips it),
  // pl pad columns feed zero P values into the K=32 WMMA.
  if constexpr (NK > NWIN) {
    for (int i = lane; i < 64 * (NK - NWIN); i += 32) {
      int d = i / (NK - NWIN), c = NWIN + i % (NK - NWIN);
      if (wiw == 0) vt[d * NK + c] = 0;
    }
    for (int i = lane; i < 16 * (NK - NWIN); i += 32) {
      int rr = i / (NK - NWIN), cc = NWIN + i % (NK - NWIN);
      pl[rr * NK + cc] = 0;
    }
  }

  // one wave per window DMAs the transposed V tile into LDS via the TDM
  if (wiw == 0) {
    const u16* gsrc =
        vtg + ((b * 3 + head - 1) * 64) * HWSZ + (wy * WS) * WIDTH + wx * WS;
    tdm_load_vt<WS, (NK > NWIN)>((unsigned)(winLocal * VT_W * 2), gsrc);
    __builtin_amdgcn_s_wait_tensorcnt(0);
  }
  __syncthreads();

  for (int it = 0; it < QTPW; ++it) {
    int q0 = (wiw * QTPW + it) * 16;
    int qtok = q0 + ln;
    const u16* qrow =
        qkv + (base + (qtok / WS) * WIDTH + (qtok % WS)) * 768 + qoff;
    v16bf a0 = load_frag_a(qrow, h);
    v16bf a1 = load_frag_a(qrow + 32, h);

    // S = Q K^T  (Q already scaled)
    v8f s[NT] = {};
#pragma unroll
    for (int j = 0; j < NT; ++j) {
      int ktok = j * 16 + ln;
      const u16* krow =
          qkv + (base + (ktok / WS) * WIDTH + (ktok % WS)) * 768 + koff;
      v16bf b0 = load_frag_b(krow, h);
      v16bf b1 = load_frag_b(krow + 32, h);
      s[j] = wmma_bf16(a0, b0, s[j]);
      s[j] = wmma_bf16(a1, b1, s[j]);
    }

    // softmax: row m = h*8+r lives in one 16-lane half (xor masks 1..8)
    float rinv[8];
#pragma unroll
    for (int r = 0; r < 8; ++r) {
      float m = s[0][r];
#pragma unroll
      for (int j = 1; j < NT; ++j) m = fmaxf(m, s[j][r]);
#pragma unroll
      for (int xm = 1; xm < 16; xm <<= 1) m = fmaxf(m, __shfl_xor(m, xm, 32));
      float sum = 0.f;
#pragma unroll
      for (int j = 0; j < NT; ++j) {
        float e = __expf(s[j][r] - m);
        s[j][r] = e;
        sum += e;
      }
#pragma unroll
      for (int xm = 1; xm < 16; xm <<= 1) sum += __shfl_xor(sum, xm, 32);
      rinv[r] = 1.0f / sum;
    }

    // stage P (bf16) to LDS in A-loadable row-major form
#pragma unroll
    for (int j = 0; j < NT; ++j)
#pragma unroll
      for (int r = 0; r < 8; ++r)
        pl[(h * 8 + r) * NK + j * 16 + ln] = f2bf(s[j][r] * rinv[r]);

    // O = P V
    v8f o[4] = {};
#pragma unroll
    for (int kc = 0; kc < NK / 32; ++kc) {
      v16bf pa = load_frag_a(pl + ln * NK + kc * 32, h);
#pragma unroll
      for (int dt = 0; dt < 4; ++dt) {
        v16bf vb = load_frag_b(vt + (dt * 16 + ln) * NK + kc * 32, h);
        o[dt] = wmma_bf16(pa, vb, o[dt]);
      }
    }

    // store O: lane has fixed d column, rows h*8+r
#pragma unroll
    for (int dt = 0; dt < 4; ++dt) {
#pragma unroll
      for (int r = 0; r < 8; ++r) {
        int tok = q0 + h * 8 + r;
        int pix = base + (tok / WS) * WIDTH + (tok % WS);
        y[pix * DIMC + head * HD + dt * 16 + ln] = f2bf(o[dt][r]);
      }
    }
  }
}

// ---------------- ws=2 head (4-token windows): plain VALU ---------------------
__global__ __launch_bounds__(256) void attn_ws2(const u16* __restrict__ qkv,
                                                u16* __restrict__ y) {
  int p = blockIdx.x * 256 + threadIdx.x;
  int hw = p & (HWSZ - 1);
  int row = hw >> 7, col = hw & 127;
  int base = (p - hw) + (row & ~1) * WIDTH + (col & ~1);
  const u16* q = qkv + p * 768;
  float s[4];
#pragma unroll
  for (int j = 0; j < 4; ++j) {
    const u16* k = qkv + (base + (j >> 1) * WIDTH + (j & 1)) * 768 + DIMC;
    float acc = 0.f;
#pragma unroll 8
    for (int d = 0; d < HD; ++d) acc += bf2f(q[d]) * bf2f(k[d]);
    s[j] = acc;
  }
  float m = fmaxf(fmaxf(s[0], s[1]), fmaxf(s[2], s[3]));
  float sum = 0.f;
#pragma unroll
  for (int j = 0; j < 4; ++j) { s[j] = __expf(s[j] - m); sum += s[j]; }
  float inv = 1.0f / sum;
#pragma unroll 8
  for (int d = 0; d < HD; ++d) {
    float o = 0.f;
#pragma unroll
    for (int j = 0; j < 4; ++j)
      o += s[j] * bf2f(qkv[(base + (j >> 1) * WIDTH + (j & 1)) * 768 + 2 * DIMC + d]);
    y[p * DIMC + d] = f2bf(o * inv);
  }
}

// ---------------- proj GEMM + bias + residual ---------------------------------
__global__ __launch_bounds__(256) void proj_gemm(const u16* __restrict__ yb,
                                                 const u16* __restrict__ wp,
                                                 const float* __restrict__ bias,
                                                 const float* __restrict__ x,
                                                 float* __restrict__ out) {
  int wave = threadIdx.x >> 5, lane = threadIdx.x & 31;
  int h = lane >> 4, ln = lane & 15;
  int p0 = blockIdx.x * 128 + wave * 16;
  int o0 = blockIdx.y * 64;
  const u16* arow = yb + (p0 + ln) * DIMC;
  __builtin_prefetch(arow + 128, 0, 2);
  v8f acc[4] = {};
#pragma unroll
  for (int k0 = 0; k0 < DIMC; k0 += 32) {
    v16bf a = load_frag_a(arow + k0, h);
#pragma unroll
    for (int t = 0; t < 4; ++t) {
      v16bf b = load_frag_b(wp + (o0 + t * 16 + ln) * DIMC + k0, h);
      acc[t] = wmma_bf16(a, b, acc[t]);
    }
  }
  int bidx = p0 >> 14;
  int hw = (p0 & (HWSZ - 1)) + h * 8;
#pragma unroll
  for (int t = 0; t < 4; ++t) {
    int o = o0 + t * 16 + ln;
    float bv = bias[o];
    int basei = (bidx * DIMC + o) * HWSZ + hw;
    const float* xr = x + basei;
    v4f r0, r1;
#pragma unroll
    for (int r = 0; r < 4; ++r) {
      r0[r] = acc[t][r] + bv + xr[r];
      r1[r] = acc[t][r + 4] + bv + xr[r + 4];
    }
    *(v4f*)(out + basei) = r0;
    *(v4f*)(out + basei + 4) = r1;
  }
}

extern "C" void kernel_launch(void* const* d_in, const int* in_sizes, int n_in,
                              void* d_out, int out_size, void* d_ws, size_t ws_size,
                              hipStream_t stream) {
  const float* x      = (const float*)d_in[0];
  const float* qkv_w  = (const float*)d_in[1];
  const float* proj_w = (const float*)d_in[2];
  const float* proj_b = (const float*)d_in[3];
  float* out = (float*)d_out;

  char* ws = (char*)d_ws;
  const size_t NPIX = (size_t)BATCH * HWSZ;                 // 131072
  size_t off = 0;
  u16* xt   = (u16*)(ws + off); off += NPIX * DIMC * 2;     // 64 MB
  u16* qkvb = (u16*)(ws + off); off += NPIX * 768 * 2;      // 192 MB
  u16* yb   = (u16*)(ws + off); off += NPIX * DIMC * 2;     // 64 MB
  u16* wqb  = (u16*)(ws + off); off += 768 * DIMC * 2;
  u16* wpb  = (u16*)(ws + off); off += DIMC * DIMC * 2;
  u16* vtg  = (u16*)(ws + off); off += (size_t)BATCH * 3 * 64 * HWSZ * 2; // 48 MB

  // convert weights + transpose/convert activations
  cvt_bf16<<<(768 * DIMC + 255) / 256, 256, 0, stream>>>(qkv_w, wqb, 768 * DIMC);
  cvt_bf16<<<(DIMC * DIMC + 255) / 256, 256, 0, stream>>>(proj_w, wpb, DIMC * DIMC);
  xpose_cvt<<<dim3(HWSZ / 32, DIMC / 32, BATCH), dim3(32, 8), 0, stream>>>(x, xt);

  // QKV projection (bf16 WMMA, SCALE folded into Q)
  qkv_gemm<<<dim3((unsigned)(NPIX / 512), 48), 256, 0, stream>>>(xt, wqb, qkvb);

  // per-head transposed V planes for TDM tiles
  vsplit<<<dim3(HWSZ / 32, 3, BATCH), 256, 0, stream>>>(qkvb, vtg);

  // attention per head
  attn_ws2<<<(unsigned)(NPIX / 256), 256, 0, stream>>>(qkvb, yb);          // ws=2
  attn_wmma<4, 8, 8><<<1024, 256, 0, stream>>>(qkvb, vtg, yb, 1);          // ws=4
  attn_wmma<8, 8, 2><<<1024, 256, 0, stream>>>(qkvb, vtg, yb, 2);          // ws=8
  attn_wmma<16, 2, 1><<<512, 64, 0, stream>>>(qkvb, vtg, yb, 3);           // ws=16

  // output projection + bias + residual
  proj_gemm<<<dim3((unsigned)(NPIX / 128), 4), 256, 0, stream>>>(yb, wpb, proj_b, x, out);
}
